// EdgeDecoder_9534827397910
// MI455X (gfx1250) — matrix operand order repaired and checked by reading, not compiled, not run on hardware
//
#include <hip/hip_runtime.h>

typedef __attribute__((ext_vector_type(16))) _Float16 v16h;
typedef __attribute__((ext_vector_type(8)))  _Float16 v8h;
typedef __attribute__((ext_vector_type(4)))  _Float16 v4h;
typedef __attribute__((ext_vector_type(8)))  float    v8f;

static constexpr int WAVES   = 8;     // waves per block (wave32)
static constexpr int BLOCK_T = WAVES * 32;
static constexpr int EPW     = 32;    // edges per wave tile (2 WMMA M-tiles)
static constexpr int EPB     = WAVES * EPW; // 256 edges per block iteration

// Padded LDS row strides (f16 elements): odd multiple of 4 dwords so that
// successive rows shift 4 banks -> conflict-free b128 fragment loads.
static constexpr int ASTR = 264;      // activation rows (256 cols + 8 pad), 528B
static constexpr int WSTR1 = 264;     // W1^T rows (256 cols + 8 pad), 528B
static constexpr int WSTR  = 136;     // W2..W4^T rows (128 cols + 8 pad), 272B

// ---- LDS byte layout (dynamic shared; 320KB/WGP on CDNA5) ----
static constexpr int W1T_OFF  = 0;                        // 128 x 264 f16
static constexpr int W2T_OFF  = W1T_OFF + 128*WSTR1*2;    // 128 x 136 f16
static constexpr int W3T_OFF  = W2T_OFF + 128*WSTR*2;
static constexpr int W4T_OFF  = W3T_OFF + 128*WSTR*2;
static constexpr int BIAS_OFF = W4T_OFF + 128*WSTR*2;     // 4 x 128 f32
static constexpr int W5_OFF   = BIAS_OFF + 4*128*4;       // 128 f32
static constexpr int ACT_OFF  = W5_OFF + 128*4;           // 8 waves x 32 x 264 f16
static constexpr int ACT_WAVE = EPW * ASTR * 2;           // bytes per wave (16B aligned)
static constexpr int LDS_BYTES = ACT_OFF + WAVES*ACT_WAVE; // ~302.5 KB

// One MLP layer for a wave's 32-edge tile (two 16-row M-tiles sharing B fragments).
// sAct: [32][ASTR] f16, input in cols 0..KD-1; output (bias+relu, f16) to cols 0..127.
// sWt:  W^T [128][WS] f16 (row = output channel, contiguous K).
template<int KD, int WS>
__device__ __forceinline__ void mlp_layer(_Float16* __restrict__ sAct,
                                          const _Float16* __restrict__ sWt,
                                          const float* __restrict__ sBias,
                                          int lane) {
    const int l16 = lane & 15;
    const int lh  = lane >> 4;          // 0/1: which lane half
    const v8f vzero = {0.f,0.f,0.f,0.f,0.f,0.f,0.f,0.f};
    v8f acc0[8], acc1[8];
#pragma unroll
    for (int nt = 0; nt < 8; ++nt) { acc0[nt] = vzero; acc1[nt] = vzero; }

#pragma unroll
    for (int kt = 0; kt < KD/32; ++kt) {
        // A fragments 16x32 f16: lane holds row M=l16 (tile0) / 16+l16 (tile1);
        // K = base..base+7 and base+16..base+23 (base = kt*32 + lh*8)
        const _Float16* ap0 = sAct + l16*ASTR + kt*32 + lh*8;
        const _Float16* ap1 = ap0 + 16*ASTR;
        v8h a0lo = *(const v8h*)ap0;
        v8h a0hi = *(const v8h*)(ap0 + 16);
        v8h a1lo = *(const v8h*)ap1;
        v8h a1hi = *(const v8h*)(ap1 + 16);
        v16h a0 = __builtin_shufflevector(a0lo, a0hi,
                  0,1,2,3,4,5,6,7,8,9,10,11,12,13,14,15);
        v16h a1 = __builtin_shufflevector(a1lo, a1hi,
                  0,1,2,3,4,5,6,7,8,9,10,11,12,13,14,15);
#pragma unroll
        for (int nt = 0; nt < 8; ++nt) {
            // B fragment 32x16 f16: lane holds col N=nt*16+l16, K = base..base+7 and base+16..base+23 (base = kt*32 + lh*8)
            const _Float16* bp = sWt + (nt*16 + l16)*WS + kt*32 + lh*8;
            v8h blo = *(const v8h*)bp;
            v8h bhi = *(const v8h*)(bp + 16);
            v16h b = __builtin_shufflevector(blo, bhi,
                     0,1,2,3,4,5,6,7,8,9,10,11,12,13,14,15);
            // one B fragment feeds two WMMAs (2x arithmetic intensity vs LDS)
            acc0[nt] = __builtin_amdgcn_wmma_f32_16x16x32_f16(
                false, a0, false, b, (short)0, acc0[nt], false, false);
            acc1[nt] = __builtin_amdgcn_wmma_f32_16x16x32_f16(
                false, a1, false, b, (short)0, acc1[nt], false, false);
        }
    }
    // bias + relu, write back f16 to cols 0..127.
    // C/D layout: VGPR r -> row M = r + 8*lh (+16 for tile1), col N = nt*16 + l16
#pragma unroll
    for (int nt = 0; nt < 8; ++nt) {
        float bv = sBias[nt*16 + l16];
#pragma unroll
        for (int r = 0; r < 8; ++r) {
            float x0 = acc0[nt][r] + bv;
            float x1 = acc1[nt][r] + bv;
            x0 = x0 > 0.f ? x0 : 0.f;
            x1 = x1 > 0.f ? x1 : 0.f;
            sAct[(r + lh*8)*ASTR      + nt*16 + l16] = (_Float16)x0;
            sAct[(r + lh*8 + 16)*ASTR + nt*16 + l16] = (_Float16)x1;
        }
    }
}

// LDS (~302KB) limits us to one 8-wave block per WGP (2 waves/SIMD),
// so release the VGPR budget: min-waves-per-EU = 1 -> no 128-VGPR cap, no spills.
__global__ void __launch_bounds__(BLOCK_T, 1) edge_decoder_kernel(
    const float* __restrict__ zA, const float* __restrict__ zT,
    const long long* __restrict__ eli,
    const float* __restrict__ W1, const float* __restrict__ b1,
    const float* __restrict__ W2, const float* __restrict__ b2,
    const float* __restrict__ W3, const float* __restrict__ b3,
    const float* __restrict__ W4, const float* __restrict__ b4,
    const float* __restrict__ W5, const float* __restrict__ b5,
    float* __restrict__ out, long long E)
{
    extern __shared__ char smem[];
    _Float16* sW1t = (_Float16*)(smem + W1T_OFF);
    _Float16* sW2t = (_Float16*)(smem + W2T_OFF);
    _Float16* sW3t = (_Float16*)(smem + W3T_OFF);
    _Float16* sW4t = (_Float16*)(smem + W4T_OFF);
    float*    sBias = (float*)(smem + BIAS_OFF);
    float*    sW5   = (float*)(smem + W5_OFF);

    const int tid = threadIdx.x;

    // ---- stage weights once per block: fp32 -> f16, transposed (W^T[n][k]) ----
    for (int i = tid; i < 256*128; i += BLOCK_T) {
        int k = i >> 7, n = i & 127;
        sW1t[n*WSTR1 + k] = (_Float16)W1[i];
    }
    for (int i = tid; i < 128*128; i += BLOCK_T) {
        int k = i >> 7, n = i & 127;
        sW2t[n*WSTR + k] = (_Float16)W2[i];
        sW3t[n*WSTR + k] = (_Float16)W3[i];
        sW4t[n*WSTR + k] = (_Float16)W4[i];
    }
    if (tid < 128) {
        sBias[      tid] = b1[tid];
        sBias[128 + tid] = b2[tid];
        sBias[256 + tid] = b3[tid];
        sBias[384 + tid] = b4[tid];
        sW5[tid] = W5[tid];
    }
    __syncthreads();

    const int wave = tid >> 5;
    const int lane = tid & 31;
    const int l16  = lane & 15;
    const int lh   = lane >> 4;
    _Float16* sAct = (_Float16*)(smem + ACT_OFF + wave * ACT_WAVE);
    const float4* zA4 = (const float4*)zA;
    const float4* zT4 = (const float4*)zT;
    const float b5v = b5[0];

    const long long blkStride = (long long)gridDim.x * EPB;
    for (long long bb = (long long)blockIdx.x * EPB; bb < E; bb += blkStride) {
        // ---- gather: 32 edges per wave, concat [author|topic] as f16 rows [32][256] ----
        const long long wbase = bb + (long long)wave * EPW;
        const long long egl   = wbase + lane;
        int rIdx = 0, cIdx = 0;
        if (egl < E) {
            rIdx = (int)eli[egl];        // row = edge_label_index[0][e]
            cIdx = (int)eli[E + egl];    // col = edge_label_index[1][e]
        }
#pragma unroll 4
        for (int e = 0; e < EPW; ++e) {
            int r = __shfl(rIdx, e, 32);
            int c = __shfl(cIdx, e, 32);
            float4 av = zA4[(long long)r*32 + lane];   // 32 lanes x float4 = 128 floats
            float4 tv = zT4[(long long)c*32 + lane];
            v4h ah = {(_Float16)av.x, (_Float16)av.y, (_Float16)av.z, (_Float16)av.w};
            v4h th = {(_Float16)tv.x, (_Float16)tv.y, (_Float16)tv.z, (_Float16)tv.w};
            *(v4h*)(sAct + e*ASTR +       lane*4) = ah;
            *(v4h*)(sAct + e*ASTR + 128 + lane*4) = th;
        }
        __syncthreads();

        mlp_layer<256, WSTR1>(sAct, sW1t, sBias + 0,   lane);
        __syncthreads();
        mlp_layer<128, WSTR >(sAct, sW2t, sBias + 128, lane);
        __syncthreads();
        mlp_layer<128, WSTR >(sAct, sW3t, sBias + 256, lane);
        __syncthreads();
        mlp_layer<128, WSTR >(sAct, sW4t, sBias + 384, lane);
        __syncthreads();

        // ---- final 128 -> 1 dot for 32 edges; lane halves split K, shfl_xor(16) reduce ----
        const float* w5p = sW5 + lh*64;
#pragma unroll
        for (int half = 0; half < 2; ++half) {
            const int m = half*16 + l16;
            const _Float16* arow = sAct + m*ASTR + lh*64;
            float sum = 0.f;
#pragma unroll
            for (int kb = 0; kb < 8; ++kb) {
                v8h hv = *(const v8h*)(arow + kb*8);
                float4 w0 = *(const float4*)(w5p + kb*8);
                float4 w1 = *(const float4*)(w5p + kb*8 + 4);
                sum += (float)hv[0]*w0.x + (float)hv[1]*w0.y
                     + (float)hv[2]*w0.z + (float)hv[3]*w0.w
                     + (float)hv[4]*w1.x + (float)hv[5]*w1.y
                     + (float)hv[6]*w1.z + (float)hv[7]*w1.w;
            }
            sum += __shfl_xor(sum, 16, 32);
            const long long eo = wbase + m;
            if (lh == 0 && eo < E) out[eo] = sum + b5v;
        }
        __syncthreads();
    }
}

extern "C" void kernel_launch(void* const* d_in, const int* in_sizes, int n_in,
                              void* d_out, int out_size, void* d_ws, size_t ws_size,
                              hipStream_t stream) {
    const float*     zA  = (const float*)d_in[0];
    const float*     zT  = (const float*)d_in[1];
    const long long* eli = (const long long*)d_in[2];   // int64 indices
    const float* W1 = (const float*)d_in[3];
    const float* b1 = (const float*)d_in[4];
    const float* W2 = (const float*)d_in[5];
    const float* b2 = (const float*)d_in[6];
    const float* W3 = (const float*)d_in[7];
    const float* b3 = (const float*)d_in[8];
    const float* W4 = (const float*)d_in[9];
    const float* b4 = (const float*)d_in[10];
    const float* W5 = (const float*)d_in[11];
    const float* b5 = (const float*)d_in[12];
    float* out = (float*)d_out;

    const long long E = (long long)(in_sizes[2] / 2);
    int tiles = (int)((E + EPB - 1) / EPB);
    int grid  = tiles < 512 ? tiles : 512;   // amortize weight stage over many tiles/block

    (void)hipFuncSetAttribute((const void*)edge_decoder_kernel,
                              hipFuncAttributeMaxDynamicSharedMemorySize, LDS_BYTES);
    edge_decoder_kernel<<<dim3(grid), dim3(BLOCK_T), LDS_BYTES, stream>>>(
        zA, zT, eli, W1, b1, W2, b2, W3, b3, W4, b4, W5, b5, out, E);
}
